// AttentionMoEQKVSeperate_89189290869041
// MI455X (gfx1250) — matrix operand order repaired and checked by reading, not compiled, not run on hardware
//
#include <hip/hip_runtime.h>
#include <hip/hip_bf16.h>

#define N_TOK 8192
#define DIM   1024
#define NEXP  16
#define KSEL  2
#define CAP   1024
#define NSLOT (N_TOK*KSEL)

// GEMM tiling
#define MT      128          // rows per block tile
#define KCH     32           // K chunk staged in LDS per step (= WMMA K)
#define FPITCH  36           // raw f32 LDS tile pitch (floats)
#define BPITCH  72           // bf16 LDS tile pitch (elements): conflict-free frags
#define NCHUNK  (DIM / KCH)  // 32

typedef __attribute__((ext_vector_type(16))) __bf16 v16bf;
typedef __attribute__((ext_vector_type(8)))  __bf16 v8bf;
typedef __attribute__((ext_vector_type(8)))  float  v8f;
typedef int v4i_ls __attribute__((vector_size(4 * sizeof(int))));

#if defined(__has_builtin)
#  if __has_builtin(__builtin_amdgcn_global_load_async_to_lds_b128)
#    define HAVE_ASYNC_LDS 1
#  endif
#  if __has_builtin(__builtin_amdgcn_s_wait_asynccnt)
#    define HAVE_WAIT_BUILTIN 1
#  endif
#endif

#if defined(HAVE_ASYNC_LDS)
#  if defined(HAVE_WAIT_BUILTIN)
#    define WAIT_ASYNC() __builtin_amdgcn_s_wait_asynccnt(0)
#  else
#    define WAIT_ASYNC() asm volatile("s_wait_asynccnt 0" ::: "memory")
#  endif
#else
#  define WAIT_ASYNC() ((void)0)
#endif

// ---------------- kernel 0: zero per-expert fill counters --------------------
__global__ void moe_zero_fill(int* __restrict__ fill) {
  int t = threadIdx.x;
  if (t < NEXP) fill[t] = 0;
}

// ---------------- kernel 1: gate (logits -> top2 -> softmax2) ----------------
__global__ void moe_gate(const float* __restrict__ inp,
                         const float* __restrict__ gw,
                         const float* __restrict__ gb,
                         int*   __restrict__ eid,
                         float* __restrict__ sw,
                         float* __restrict__ tsc) {
  int n = blockIdx.x * blockDim.x + threadIdx.x;
  if (n >= N_TOK) return;
  float acc[NEXP];
  #pragma unroll
  for (int e = 0; e < NEXP; ++e) acc[e] = gb[e];
  const float* x = inp + (size_t)n * DIM;
  for (int d = 0; d < DIM; ++d) {
    float xv = x[d];
    const float* w = gw + (size_t)d * NEXP;   // [D,E] row-major
    #pragma unroll
    for (int e = 0; e < NEXP; ++e) acc[e] = fmaf(xv, w[e], acc[e]);
  }
  float v0 = -__builtin_inff(); int i0 = 0;
  float v1 = -__builtin_inff(); int i1 = 0;
  #pragma unroll
  for (int e = 0; e < NEXP; ++e) {
    float v = acc[e];
    if (v > v0)      { v1 = v0; i1 = i0; v0 = v; i0 = e; }
    else if (v > v1) { v1 = v;  i1 = e; }
  }
  float t   = expf(v1 - v0);
  float inv = 1.0f / (1.0f + t);
  float s0 = inv, s1 = t * inv;
  eid[n*2+0] = i0;  eid[n*2+1] = i1;
  sw [n*2+0] = s0;  sw [n*2+1] = s1;
  tsc[n] = s0;
}

// ---------------- kernel 2: capacity keep-mask + per-expert compaction -------
__global__ void moe_keep(const int*   __restrict__ eid,
                         const float* __restrict__ tsc,
                         int* __restrict__ keep,
                         int* __restrict__ fill,
                         int* __restrict__ list) {
  int gid  = blockIdx.x * blockDim.x + threadIdx.x;
  int slot = gid >> 4;
  int part = gid & 15;
  int e    = eid[slot];
  float sc = tsc[slot >> 1];
  int rank = 0;
  const int span = NSLOT / 16;
  int begin = part * span;
  for (int t = begin; t < begin + span; ++t) {
    if (eid[t] == e) {
      float ts = tsc[t >> 1];
      if (ts > sc || (ts == sc && t < slot)) ++rank;
    }
  }
  #pragma unroll
  for (int off = 1; off < 16; off <<= 1)
    rank += __shfl_xor(rank, off, 32);
  if (part == 0) {
    int k = (rank < CAP) ? 1 : 0;
    keep[slot] = k;
    if (k) {
      int pos = atomicAdd(&fill[e], 1);
      list[e * CAP + pos] = slot;
    }
  }
}

// ---------------- kernel 3: init out with dropped-slot pass-through ----------
__global__ void moe_init_out(const float* __restrict__ inp,
                             const int*   __restrict__ keep,
                             const float* __restrict__ sw,
                             float* __restrict__ out) {
  int gid = blockIdx.x * blockDim.x + threadIdx.x;
  int idx = gid * 4;
  int n = idx >> 10;
  float w = 0.0f;
  if (!keep[n*2+0]) w += sw[n*2+0];
  if (!keep[n*2+1]) w += sw[n*2+1];
  const float4 v = *(const float4*)(inp + idx);
  float4 r; r.x = w*v.x; r.y = w*v.y; r.z = w*v.z; r.w = w*v.w;
  *(float4*)(out + idx) = r;
}

// ---------------- kernel 4: per-expert GEMM, bf16-split WMMA, async LDS ------
// Block: 8 waves, tile = 128 rows x 128 cols. A is async-copied (raw f32) into
// LDS, converted ONCE per block into bf16 hi/lo tiles, then each wave ds-loads
// ready-made A fragments and runs 3 WMMAs per row-tile (hi*hi + hi*lo + lo*hi).
__global__ __launch_bounds__(256) void moe_expert_gemm(
    const float* __restrict__ inp, const float* __restrict__ ew,
    const float* __restrict__ eb,  const int*   __restrict__ list,
    const int*   __restrict__ fill, const float* __restrict__ sw,
    float* __restrict__ out) {
  const int e    = blockIdx.y >> 3;       // CAP/MT = 8 row-tiles per expert
  const int tile = blockIdx.y & 7;
  const int cnt  = fill[e];
  const int row0 = tile * MT;
  if (row0 >= cnt) return;                // uniform: EXEC all-ones at WMMA

  __shared__ float    ftile[2][MT * FPITCH];   // raw f32, double-buffered
  __shared__ __bf16   ahtile[MT * BPITCH];     // bf16 hi
  __shared__ __bf16   altile[MT * BPITCH];     // bf16 lo
  __shared__ int      rowSlot[MT];
  __shared__ float    rowWt[MT];
  __shared__ unsigned rowOffB[MT];

  const int tid  = threadIdx.x;
  const int wave = tid >> 5;
  const int lane = tid & 31;
  const int half = lane >> 4;
  const int nl   = lane & 15;
  const int col  = blockIdx.x * 128 + wave * 16 + nl;

  if (tid < MT) {
    int ridx = row0 + tid;
    int s = list[e * CAP + ((ridx < cnt) ? ridx : row0)];
    rowSlot[tid] = s;
    rowWt[tid]   = sw[s];
    rowOffB[tid] = (unsigned)(s >> 1) * (unsigned)(DIM * 4);
  }
  __syncthreads();

  const char*  inp_c = (const char*)inp;
  const float* brow  = ew + ((size_t)e * DIM + col) * DIM;   // row `col` of W_e

  v8f acc[8];
  #pragma unroll
  for (int rt = 0; rt < 8; ++rt) { v8f z = {}; acc[rt] = z; }

  // async-stage one K-chunk (MT rows x KCH floats) into ftile[b]
  auto stage = [&](int kk, int b) {
    #pragma unroll
    for (int i = 0; i < (MT * KCH / 4) / 256; ++i) {   // 4 x b128 per thread
      int id  = i * 256 + tid;
      int row = id >> 3;                  // KCH/4 = 8 segments per row
      int seg = id & 7;
      unsigned goff = rowOffB[row] + (unsigned)(kk + seg * 4) * 4u;
      float* ldst = &ftile[b][row * FPITCH + seg * 4];
#if defined(HAVE_ASYNC_LDS)
      __builtin_amdgcn_global_load_async_to_lds_b128(
          (v4i_ls*)(inp_c + goff), (v4i_ls*)ldst, 0, 0);
#else
      const float4 v = *(const float4*)(inp_c + goff);
      *(float4*)ldst = v;
#endif
    }
  };

  stage(0, 0);
  for (int c = 0; c < NCHUNK; ++c) {
    WAIT_ASYNC();            // own chunk-c copy done
    __syncthreads();         // all copies done; all prior-iter LDS reads done
    if (c + 1 < NCHUNK) stage((c + 1) * KCH, (c + 1) & 1);

    // one-pass f32 -> bf16(hi/lo) conversion: 16 elements per thread
    {
      const float* fr = &ftile[c & 1][(tid >> 1) * FPITCH + (tid & 1) * 16];
      __bf16* dh = ahtile + (tid >> 1) * BPITCH + (tid & 1) * 16;
      __bf16* dl = altile + (tid >> 1) * BPITCH + (tid & 1) * 16;
      v8bf h0, h1, l0, l1;
      #pragma unroll
      for (int j = 0; j < 8; ++j) {
        float f = fr[j];
        __bf16 h = (__bf16)f;
        h0[j] = h; l0[j] = (__bf16)(f - (float)h);
      }
      #pragma unroll
      for (int j = 0; j < 8; ++j) {
        float f = fr[8 + j];
        __bf16 h = (__bf16)f;
        h1[j] = h; l1[j] = (__bf16)(f - (float)h);
      }
      *(v8bf*)(dh)     = h0;  *(v8bf*)(dh + 8) = h1;
      *(v8bf*)(dl)     = l0;  *(v8bf*)(dl + 8) = l1;
    }
    __syncthreads();         // bf16 tiles ready for all waves

    const int kkg = c * KCH;
    // B fragment (32x16 bf16): half 0 -> K [0..8)+[16..24), half 1 -> +8
    v16bf bhi, blo;
    #pragma unroll
    for (int j = 0; j < 16; ++j) {
      float f = brow[kkg + (j >> 3) * 16 + half * 8 + (j & 7)];
      __bf16 h = (__bf16)f;
      bhi[j] = h; blo[j] = (__bf16)(f - (float)h);
    }
    #pragma unroll
    for (int rt = 0; rt < 8; ++rt) {
      // A fragment (16x32 bf16): half 0 -> K [0..8)+[16..24), half 1 -> +8
      const __bf16* ah = ahtile + (rt * 16 + nl) * BPITCH;
      const __bf16* al = altile + (rt * 16 + nl) * BPITCH;
      v8bf a0 = *(const v8bf*)(ah + half * 8);
      v8bf a1 = *(const v8bf*)(ah + 16 + half * 8);
      v8bf b0 = *(const v8bf*)(al + half * 8);
      v8bf b1 = *(const v8bf*)(al + 16 + half * 8);
      v16bf ahi = __builtin_shufflevector(a0, a1, 0,1,2,3,4,5,6,7,8,9,10,11,12,13,14,15);
      v16bf alo = __builtin_shufflevector(b0, b1, 0,1,2,3,4,5,6,7,8,9,10,11,12,13,14,15);
      acc[rt] = __builtin_amdgcn_wmma_f32_16x16x32_bf16(false, ahi, false, bhi, (short)0, acc[rt], false, false);
      acc[rt] = __builtin_amdgcn_wmma_f32_16x16x32_bf16(false, ahi, false, blo, (short)0, acc[rt], false, false);
      acc[rt] = __builtin_amdgcn_wmma_f32_16x16x32_bf16(false, alo, false, bhi, (short)0, acc[rt], false, false);
    }
  }

  float bias = eb[e * DIM + col];
  #pragma unroll
  for (int rt = 0; rt < 8; ++rt) {
    #pragma unroll
    for (int r = 0; r < 8; ++r) {
      int m = rt * 16 + half * 8 + r;     // C/D layout: VGPR r -> M = r + half*8
      if (row0 + m < cnt) {
        int s   = rowSlot[m];
        float w = rowWt[m];
        atomicAdd(out + (size_t)(s >> 1) * DIM + col, w * (acc[rt][r] + bias));
      }
    }
  }
}

extern "C" void kernel_launch(void* const* d_in, const int* in_sizes, int n_in,
                              void* d_out, int out_size, void* d_ws, size_t ws_size,
                              hipStream_t stream) {
  (void)in_sizes; (void)n_in; (void)out_size; (void)ws_size;
  const float* inp = (const float*)d_in[0];   // [N, D]
  const float* gw  = (const float*)d_in[1];   // [D, E]
  const float* gb  = (const float*)d_in[2];   // [E]
  const float* ew  = (const float*)d_in[3];   // [E, D, D]
  const float* eb  = (const float*)d_in[4];   // [E, D]
  float* out = (float*)d_out;                 // [N, D]

  char* ws = (char*)d_ws;
  int*   eid  = (int*)  (ws + 0);        // [NSLOT]
  float* sw   = (float*)(ws + 65536);    // [NSLOT]
  float* tsc  = (float*)(ws + 131072);   // [N_TOK]
  int*   keep = (int*)  (ws + 163840);   // [NSLOT]
  int*   fill = (int*)  (ws + 229376);   // [NEXP]
  int*   list = (int*)  (ws + 229632);   // [NEXP*CAP]

  moe_zero_fill<<<1, 32, 0, stream>>>(fill);
  moe_gate<<<N_TOK / 256, 256, 0, stream>>>(inp, gw, gb, eid, sw, tsc);
  moe_keep<<<(NSLOT * 16) / 256, 256, 0, stream>>>(eid, tsc, keep, fill, list);
  moe_init_out<<<(N_TOK * DIM / 4) / 256, 256, 0, stream>>>(inp, keep, sw, out);
  moe_expert_gemm<<<dim3(DIM / 128, NEXP * (CAP / MT)), 256, 0, stream>>>(
      inp, ew, eb, list, fill, sw, out);
}
